// SelfAttention_37873021616170
// MI455X (gfx1250) — matrix operand order repaired and checked by reading, not compiled, not run on hardware
//
#include <hip/hip_runtime.h>
#include <hip/hip_bf16.h>

typedef __attribute__((ext_vector_type(16))) _Float16 v16h;
typedef __attribute__((ext_vector_type(8)))  _Float16 v8h;
typedef __attribute__((ext_vector_type(8)))  float    v8f;

#define EMBED 1024
#define HEADS 16
#define HDIM  64
#define BB    4
#define SS    2048

// ---------------------------------------------------------------------------
// Kernel 1: Wo fp32 -> f16
// ---------------------------------------------------------------------------
__global__ __launch_bounds__(256) void cvt_wo_kernel(const float* __restrict__ Wo,
                                                     _Float16* __restrict__ Woh) {
    int i = blockIdx.x * 256 + threadIdx.x;
    Woh[i] = (_Float16)Wo[i];
}

// ---------------------------------------------------------------------------
// Kernel 2: per-head QKV projections (fp32 math).
// Q,K emitted f16 in [B][H][S][64]; V emitted f16 TRANSPOSED in [B][H][64][S]
// so the attention kernel can async-copy V^T chunks with no LDS transpose.
// Q is pre-scaled by 1/sqrt(EMBED) = 1/32.
// ---------------------------------------------------------------------------
__global__ __launch_bounds__(256) void proj_kernel(
    const float* __restrict__ values, const float* __restrict__ keys,
    const float* __restrict__ query,
    const float* __restrict__ Wv, const float* __restrict__ Wk,
    const float* __restrict__ Wq,
    _Float16* __restrict__ Vpt, _Float16* __restrict__ Kp,
    _Float16* __restrict__ Qp) {
    __shared__ float WqL[64 * 64], WkL[64 * 64], WvL[64 * 64];
    __shared__ float xq[EMBED], xk[EMBED], xv[EMBED];
    const int tid = threadIdx.x;
    for (int i = tid; i < 64 * 64; i += 256) {
        WqL[i] = Wq[i]; WkL[i] = Wk[i]; WvL[i] = Wv[i];
    }
    const int token0 = blockIdx.x * 16;
    for (int tk = 0; tk < 16; ++tk) {
        const int token = token0 + tk;
        __syncthreads();
        for (int i = tid; i < EMBED; i += 256) {
            size_t off = (size_t)token * EMBED + i;
            xq[i] = query[off]; xk[i] = keys[off]; xv[i] = values[off];
        }
        __syncthreads();
        const int b = token >> 11;
        const int s = token & (SS - 1);
#pragma unroll
        for (int r = 0; r < 4; ++r) {
            const int o = r * 256 + tid;
            const int h = o >> 6;
            const int e = o & 63;
            float aq = 0.f, ak = 0.f, av = 0.f;
#pragma unroll 8
            for (int d = 0; d < 64; ++d) {
                aq += xq[h * 64 + d] * WqL[e * 64 + d];
                ak += xk[h * 64 + d] * WkL[e * 64 + d];
                av += xv[h * 64 + d] * WvL[e * 64 + d];
            }
            const size_t qk = (((size_t)b * HEADS + h) * SS + s) * 64 + e;
            Qp[qk] = (_Float16)(aq * 0.03125f);
            Kp[qk] = (_Float16)ak;
            Vpt[(((size_t)b * HEADS + h) * 64 + e) * SS + s] = (_Float16)av;
        }
    }
}

// ---------------------------------------------------------------------------
// Kernel 3: flash attention. 128 threads = 4 waves, 64 queries per block.
// 64-key chunks double-buffered in LDS via global_load_async_to_lds_b128;
// issue chunk i+1 after the barrier, compute chunk i, s_wait_asynccnt at top.
// Scores computed transposed (S^T = K x Q^T) so per-query softmax stats need
// only one shfl_xor(16) lane-pair reduction. K fragments are hoisted into
// distinct registers before the QK WMMA chain, and V^T fragments are loaded
// before the softmax so ds latency hides under the exp/max VALU work.
// ---------------------------------------------------------------------------
__global__ __launch_bounds__(128) void attn_kernel(
    const _Float16* __restrict__ Qp, const _Float16* __restrict__ Kp,
    const _Float16* __restrict__ Vpt, const int* __restrict__ mask,
    _Float16* __restrict__ ctx) {
    __shared__ __align__(16) _Float16 Kbuf[2][64][64];   // [buf][key][d]
    __shared__ __align__(16) _Float16 Vtbuf[2][64][64];  // [buf][d][key]
    __shared__ __align__(16) float    biasLds[2][64];

    const int bh   = blockIdx.y;
    const int b    = bh >> 4;
    const int h    = bh & 15;
    const int tid  = threadIdx.x;
    const int wave = tid >> 5;
    const int lane = tid & 31;
    const int n    = lane & 15;
    const int hi   = lane >> 4;

    const int qrow = blockIdx.x * 64 + wave * 16 + n;

    // Q^T B-fragments: element p <-> K = 32*c + 16*hi + p
    const _Float16* Qrow = Qp + ((size_t)bh * SS + qrow) * 64;
    const v16h bq0 = *(const v16h*)(Qrow + 16 * hi);
    const v16h bq1 = *(const v16h*)(Qrow + 32 + 16 * hi);

    const unsigned long long sbaseK =
        (unsigned long long)(Kp + (size_t)bh * SS * 64);
    const unsigned long long sbaseV =
        (unsigned long long)(Vpt + (size_t)bh * 64 * SS);
    const int* maskb = mask + b * SS;

    const unsigned kldsBase = (unsigned)(size_t)(&Kbuf[0][0][0]);
    const unsigned vldsBase = (unsigned)(size_t)(&Vtbuf[0][0][0]);

    // Issue one 64-key chunk (8KB K + 8KB V^T) as per-thread async 16B copies.
    auto issue = [&](int kcn, int buf) {
        unsigned kl = kldsBase + (unsigned)buf * 8192u + (unsigned)tid * 64u;
        unsigned ko = (unsigned)(kcn * 128 + tid * 64);
#pragma unroll
        for (int q = 0; q < 4; ++q)
            asm volatile("global_load_async_to_lds_b128 %0, %1, %2"
                         :: "v"(kl + q * 16u), "v"(ko + q * 16u), "s"(sbaseK)
                         : "memory");
        const int d = tid >> 1, half = tid & 1;
        unsigned vl = vldsBase + (unsigned)buf * 8192u +
                      (unsigned)(d * 128 + half * 64);
        unsigned vo = (unsigned)(d * (SS * 2) + kcn * 2 + half * 64);
#pragma unroll
        for (int q = 0; q < 4; ++q)
            asm volatile("global_load_async_to_lds_b128 %0, %1, %2"
                         :: "v"(vl + q * 16u), "v"(vo + q * 16u), "s"(sbaseV)
                         : "memory");
        if (tid < 64)
            biasLds[buf][tid] = (maskb[kcn + tid] == 0) ? -1.0e20f : 0.0f;
    };

    // A-fragment loader: element p <-> K = c*32 + 16*hi + p
    auto loadA = [&](const _Float16* rowptr, int c) -> v16h {
        v8h lo = *(const v8h*)(rowptr + c * 32 + 16 * hi);
        v8h hj = *(const v8h*)(rowptr + c * 32 + 16 * hi + 8);
        v16h a;
#pragma unroll
        for (int i = 0; i < 8; ++i) { a[i] = lo[i]; a[i + 8] = hj[i]; }
        return a;
    };

    v8f acc[4];
#pragma unroll
    for (int t = 0; t < 4; ++t) acc[t] = (v8f){};
    float m_run = -3.0e38f;
    float l_run = 0.0f;

    issue(0, 0);

    for (int kc = 0; kc < SS; kc += 64) {
        const int buf = (kc >> 6) & 1;
        asm volatile("s_wait_asynccnt 0x0" ::: "memory");
        __syncthreads();
        if (kc + 64 < SS) issue(kc + 64, buf ^ 1);

        // Hoist all K A-fragments into distinct registers, then WMMA chain.
        v16h ka[8];
#pragma unroll
        for (int t = 0; t < 4; ++t) {
            ka[2 * t]     = loadA(&Kbuf[buf][t * 16 + n][0], 0);
            ka[2 * t + 1] = loadA(&Kbuf[buf][t * 16 + n][0], 1);
        }
        // Bias vectors up front as well.
        v8f bt[4];
#pragma unroll
        for (int t = 0; t < 4; ++t)
            bt[t] = *(const v8f*)(&biasLds[buf][t * 16 + 8 * hi]);

        // S^T = K x Q^T : four 16-key M-tiles, two 32-wide d-chunks
        v8f cf[4];
#pragma unroll
        for (int t = 0; t < 4; ++t) {
            cf[t] = (v8f){};
            cf[t] = __builtin_amdgcn_wmma_f32_16x16x32_f16(
                false, ka[2 * t], false, bq0, (short)0, cf[t], false, false);
            cf[t] = __builtin_amdgcn_wmma_f32_16x16x32_f16(
                false, ka[2 * t + 1], false, bq1, (short)0, cf[t], false, false);
        }

        // Prefetch V^T fragments; their ds latency hides under the softmax.
        v16h va[8];
#pragma unroll
        for (int t = 0; t < 4; ++t) {
            va[2 * t]     = loadA(&Vtbuf[buf][t * 16 + n][0], 0);
            va[2 * t + 1] = loadA(&Vtbuf[buf][t * 16 + n][0], 1);
        }

        // mask bias + online softmax over 64 keys
        float smax = -3.0e38f;
#pragma unroll
        for (int t = 0; t < 4; ++t)
#pragma unroll
            for (int j = 0; j < 8; ++j) {
                cf[t][j] += bt[t][j];
                smax = fmaxf(smax, cf[t][j]);
            }
        smax = fmaxf(smax, __shfl_xor(smax, 16, 32));
        const float m_new = fmaxf(m_run, smax);
        const float scale = __expf(m_run - m_new);
        float rowsum = 0.0f;
#pragma unroll
        for (int t = 0; t < 4; ++t)
#pragma unroll
            for (int j = 0; j < 8; ++j) {
                cf[t][j] = __expf(cf[t][j] - m_new);
                rowsum += cf[t][j];
            }
        rowsum += __shfl_xor(rowsum, 16, 32);
        l_run = l_run * scale + rowsum;
        m_run = m_new;
#pragma unroll
        for (int t = 0; t < 4; ++t)
#pragma unroll
            for (int j = 0; j < 8; ++j) acc[t][j] *= scale;

        // Rebuild P^T as B fragments (lane-pair exchange, C->B layout)
        v16h bp0, bp1;
#pragma unroll
        for (int j = 0; j < 8; ++j) {
            const float o0 = __shfl_xor(cf[0][j], 16, 32);
            const float o1 = __shfl_xor(cf[1][j], 16, 32);
            const float o2 = __shfl_xor(cf[2][j], 16, 32);
            const float o3 = __shfl_xor(cf[3][j], 16, 32);
            bp0[j]     = (_Float16)(hi ? o1 : cf[0][j]);
            bp0[j + 8] = (_Float16)(hi ? cf[1][j] : o0);
            bp1[j]     = (_Float16)(hi ? o3 : cf[2][j]);
            bp1[j + 8] = (_Float16)(hi ? cf[3][j] : o2);
        }

        // Out^T += V^T x P^T : 4 d-tiles x 2 key-chunks
#pragma unroll
        for (int t = 0; t < 4; ++t) {
            acc[t] = __builtin_amdgcn_wmma_f32_16x16x32_f16(
                false, va[2 * t], false, bp0, (short)0, acc[t], false, false);
            acc[t] = __builtin_amdgcn_wmma_f32_16x16x32_f16(
                false, va[2 * t + 1], false, bp1, (short)0, acc[t], false, false);
        }
    }

    const float inv_l = 1.0f / fmaxf(l_run, 1e-30f);
    _Float16* cbase = ctx + ((size_t)b * SS + qrow) * EMBED + h * 64;
#pragma unroll
    for (int t = 0; t < 4; ++t) {
        v8h ph;
#pragma unroll
        for (int j = 0; j < 8; ++j) ph[j] = (_Float16)(acc[t][j] * inv_l);
        *(v8h*)(cbase + t * 16 + 8 * hi) = ph;
    }
}

// ---------------------------------------------------------------------------
// Kernel 4: out = ctx @ Wo^T + bo  (f16 WMMA, fp32 out)
// ---------------------------------------------------------------------------
__global__ __launch_bounds__(128) void outproj_kernel(
    const _Float16* __restrict__ ctx, const _Float16* __restrict__ Woh,
    const float* __restrict__ bo, float* __restrict__ out) {
    const int tid  = threadIdx.x;
    const int wave = tid >> 5;
    const int lane = tid & 31;
    const int n    = lane & 15;
    const int hi   = lane >> 4;
    const int rowbase = blockIdx.x * 16;
    const int col     = blockIdx.y * 64 + wave * 16 + n;

    const _Float16* arow = ctx + (size_t)(rowbase + n) * EMBED;
    const _Float16* brow = Woh + (size_t)col * EMBED;
    v8f c = (v8f){};
#pragma unroll 4
    for (int kc = 0; kc < 32; ++kc) {
        v8h lo = *(const v8h*)(arow + kc * 32 + 16 * hi);
        v8h hj = *(const v8h*)(arow + kc * 32 + 16 * hi + 8);
        v16h a;
#pragma unroll
        for (int i = 0; i < 8; ++i) { a[i] = lo[i]; a[i + 8] = hj[i]; }
        const v16h bf = *(const v16h*)(brow + kc * 32 + 16 * hi);
        c = __builtin_amdgcn_wmma_f32_16x16x32_f16(false, a, false, bf,
                                                   (short)0, c, false, false);
    }
    const float bias = bo[col];
#pragma unroll
    for (int j = 0; j < 8; ++j)
        out[(size_t)(rowbase + j + 8 * hi) * EMBED + col] = c[j] + bias;
}

// ---------------------------------------------------------------------------
extern "C" void kernel_launch(void* const* d_in, const int* in_sizes, int n_in,
                              void* d_out, int out_size, void* d_ws,
                              size_t ws_size, hipStream_t stream) {
    (void)in_sizes; (void)n_in; (void)out_size; (void)ws_size;
    const float* values = (const float*)d_in[0];
    const float* keys   = (const float*)d_in[1];
    const float* query  = (const float*)d_in[2];
    const int*   mask   = (const int*)d_in[3];
    const float* Wv     = (const float*)d_in[4];
    const float* Wk     = (const float*)d_in[5];
    const float* Wq     = (const float*)d_in[6];
    const float* Wo     = (const float*)d_in[7];
    const float* bo     = (const float*)d_in[8];
    float* out = (float*)d_out;

    _Float16* ws = (_Float16*)d_ws;
    const size_t PER = (size_t)BB * SS * EMBED;
    _Float16* Qp  = ws;
    _Float16* Kp  = Qp + PER;
    _Float16* Vpt = Kp + PER;   // transposed layout [B*H][64][S]
    _Float16* ctx = Vpt + PER;
    _Float16* Woh = ctx + PER;

    cvt_wo_kernel<<<EMBED * EMBED / 256, 256, 0, stream>>>(Wo, Woh);
    proj_kernel<<<BB * SS / 16, 256, 0, stream>>>(values, keys, query,
                                                  Wv, Wk, Wq, Vpt, Kp, Qp);
    attn_kernel<<<dim3(SS / 64, BB * HEADS), 128, 0, stream>>>(Qp, Kp, Vpt,
                                                               mask, ctx);
    outproj_kernel<<<dim3(BB * SS / 16, EMBED / 64), 128, 0, stream>>>(
        ctx, Woh, bo, out);
}
